// SSDIR_64879775973641
// MI455X (gfx1250) — matrix operand order repaired and checked by reading, not compiled, not run on hardware
//
#include <hip/hip_runtime.h>
#include <hip/hip_bf16.h>
#include <math.h>

// ---------------------------------------------------------------------------
// SSDIR decoder for MI455X (gfx1250, wave32, WMMA + Tensor Data Mover).
// Pipeline:
//   K0 convW1t / convW2t : f32 weights -> f16, transposed to N-major
//   K1 sort_gather       : idx gather (idx[j]=j>>2) + stable descending
//                          depth sort + gather what/where/pres, what -> f16
//   K2 gemm1             : (2160x64)x(64x512) WMMA f16->f32, bias+ReLU -> f16 H
//   K3 gemm2             : (2160x512)x(512x12288) WMMA; W2t K-chunks staged in
//                          LDS by TDM (tensor_load_to_lds), double buffered;
//                          bias+sigmoid -> f32 dec
//   K4 composite         : fused STN bilinear warp + presence mask +
//                          first-nonzero-along-sorted-axis composite
// ---------------------------------------------------------------------------

#define BB      4
#define N_OBJ   135
#define N_FEAT  540
#define ZWHAT   64
#define HID     512
#define OBJ     64
#define IMG     128
#define DEC_N   (3 * OBJ * OBJ)        // 12288
#define ROWS    (BB * N_FEAT)          // 2160
#define NB_COLS 512                    // columns per gemm2 block
#define KCHUNK  32                     // K per WMMA step

typedef __attribute__((ext_vector_type(16))) _Float16     v16h;
typedef __attribute__((ext_vector_type(8)))  float        v8f;
typedef __attribute__((ext_vector_type(4)))  unsigned int uint4x;
typedef __attribute__((ext_vector_type(4)))  unsigned int u32x4;
typedef __attribute__((ext_vector_type(8)))  int          i32x8;
typedef __attribute__((ext_vector_type(4)))  int          i32x4;

union FragT {
    v16h   h;
    uint4x q[2];
};

// ------------------------- weight conversion -------------------------------

__global__ void convW1t(const float* __restrict__ W1, _Float16* __restrict__ W1t) {
    int i = blockIdx.x * blockDim.x + threadIdx.x;          // over 64*512
    if (i >= ZWHAT * HID) return;
    int k = i >> 9;           // row in W1 (K)
    int n = i & 511;          // col in W1 (N)
    W1t[n * ZWHAT + k] = (_Float16)W1[i];                   // N-major
}

__global__ void convW2t(const float* __restrict__ W2, _Float16* __restrict__ W2t) {
    size_t i = (size_t)blockIdx.x * blockDim.x + threadIdx.x;  // over 512*12288
    if (i >= (size_t)HID * DEC_N) return;
    int k = (int)(i / DEC_N);
    int n = (int)(i % DEC_N);
    W2t[(size_t)n * HID + k] = (_Float16)W2[i];             // N-major
}

// ------------------------- sort + gather -----------------------------------

__global__ void sort_gather(const float* __restrict__ z_what,
                            const float* __restrict__ z_where,
                            const float* __restrict__ z_depth,
                            const int*   __restrict__ z_pres,
                            _Float16* __restrict__ s_what,
                            float*    __restrict__ s_where,
                            int*      __restrict__ s_pres) {
    int b = blockIdx.x;
    __shared__ float d[N_FEAT];
    __shared__ short ord[N_FEAT];

    for (int j = threadIdx.x; j < N_FEAT; j += blockDim.x)
        d[j] = z_depth[b * N_OBJ + (j >> 2)];               // idx[j] = j>>2
    __syncthreads();

    // stable rank for descending sort of d (ties: original order wins)
    for (int j = threadIdx.x; j < N_FEAT; j += blockDim.x) {
        float dj = d[j];
        int pos = 0;
        for (int k = 0; k < N_FEAT; ++k) {
            float dk = d[k];
            pos += (dk > dj) || ((dk == dj) && (k < j));
        }
        ord[pos] = (short)j;
    }
    __syncthreads();

    for (int p = threadIdx.x; p < N_FEAT; p += blockDim.x) {
        int j   = ord[p];
        int dst = b * N_FEAT + p;
        int src = b * N_FEAT + j;
        s_pres[dst] = z_pres[src];
#pragma unroll
        for (int c = 0; c < 4; ++c) s_where[dst * 4 + c] = z_where[src * 4 + c];
        const float* zw = z_what + (size_t)(b * N_OBJ + (j >> 2)) * ZWHAT;
        _Float16* o = s_what + (size_t)dst * ZWHAT;
#pragma unroll
        for (int k = 0; k < ZWHAT; ++k) o[k] = (_Float16)zw[k];
    }
}

// ------------------------- WMMA fragment loads -----------------------------
// A (16x32 f16, row-major src, lda in halves): lane<16 holds K 0..15,
// lane>=16 holds K 16..31; two 16B loads of contiguous halves.
__device__ __forceinline__ FragT load_a_frag(const _Float16* A, int row, int k0,
                                             int lda, int lane) {
    FragT f;
    int hi = (lane >> 4) * 16;
    const _Float16* base = A + (size_t)row * lda + k0;
    f.q[0] = *(const uint4x*)(base + hi);
    f.q[1] = *(const uint4x*)(base + 8 + hi);
    return f;
}

// B (32x16 f16) from N-major Wt (ldb = K in halves): lane holds column
// N=lane&15, K = k0 + 16*(lane>=16) + h -> 32 contiguous bytes.
__device__ __forceinline__ FragT load_b_frag(const _Float16* Wt, int n, int k0,
                                             int ldb, int lane) {
    FragT f;
    const _Float16* base = Wt + (size_t)n * ldb + k0 + ((lane >> 4) << 4);
    f.q[0] = ((const uint4x*)base)[0];
    f.q[1] = ((const uint4x*)base)[1];
    return f;
}

// B fragment from an LDS-staged K-chunk: chunk layout is [NB_COLS][KCHUNK]
// halves (row n_local = 32 contiguous halves of K).
__device__ __forceinline__ FragT load_b_frag_lds(const _Float16* chunk,
                                                 int n_local, int lane) {
    FragT f;
    const _Float16* base = chunk + n_local * KCHUNK + ((lane >> 4) << 4);
    f.q[0] = ((const uint4x*)base)[0];
    f.q[1] = ((const uint4x*)base)[1];
    return f;
}

// ------------------------- GEMM1: h = relu(what @ W1 + b1) -----------------

__global__ void gemm1(const _Float16* __restrict__ A,   // ROWS x 64
                      const _Float16* __restrict__ W1t, // 512 x 64 (N-major)
                      const float*    __restrict__ b1,
                      _Float16*       __restrict__ H) { // ROWS x 512
    int wid  = (blockIdx.x * blockDim.x + threadIdx.x) >> 5;
    int lane = threadIdx.x & 31;
    if (wid >= (ROWS / 16) * (HID / 16)) return;
    int tm = wid >> 5;          // / (HID/16)=32
    int tn = wid & 31;
    int row = tm * 16 + (lane & 15);
    int ncol = tn * 16 + (lane & 15);

    v8f acc = {};
#pragma unroll
    for (int kt = 0; kt < ZWHAT / KCHUNK; ++kt) {
        FragT a = load_a_frag(A, row, kt * KCHUNK, ZWHAT, lane);
        FragT b = load_b_frag(W1t, ncol, kt * KCHUNK, ZWHAT, lane);
        acc = __builtin_amdgcn_wmma_f32_16x16x32_f16(false, a.h, false, b.h,
                                                     (short)0, acc, false, false);
    }
    float bias  = b1[ncol];
    int   mbase = tm * 16 + ((lane >> 4) << 3);
#pragma unroll
    for (int r = 0; r < 8; ++r) {
        float v = acc[r] + bias;
        v = v > 0.f ? v : 0.f;
        H[(size_t)(mbase + r) * HID + ncol] = (_Float16)v;
    }
}

// ------------------------- TDM: W2t K-chunk -> LDS -------------------------
// 2D tensor DMA descriptor (D#): tile = KCHUNK x NB_COLS f16 elements,
// dim0 (fastest) = K with stride HID, landing contiguously in LDS
// ([NB_COLS][KCHUNK] halves = 32KB). One issue per block (wave 0), tracked
// with TENSORcnt. 6-arg builtin form (amdgpu-toolchain / clang-23):
//   (u32x4 g0, i32x8 g1, i32x4 g2, i32x4 g3, i32x8 extra, i32 cpol)
__device__ __forceinline__ void tdm_load_b_chunk(const _Float16* gsrc,
                                                 unsigned lds_off) {
    unsigned long long ga = (unsigned long long)(const void*)gsrc;
    u32x4 g0;
    g0[0] = 1u;                                          // count=1, user D#
    g0[1] = lds_off;                                     // lds_addr (bytes)
    g0[2] = (unsigned)(ga & 0xFFFFFFFFull);              // global_addr[31:0]
    g0[3] = (unsigned)((ga >> 32) & 0x1FFFFFFull)        // global_addr[56:32]
          | 0x80000000u;                                 // type = 2 ("image")
    i32x8 g1;
    g1[0] = 0x10000;                    // wg_mask=0, data_size=1 (2 bytes)
    g1[1] = (int)((unsigned)HID << 16); // tensor_dim0[15:0]=512 @ bits 63:48
    g1[2] = (int)((unsigned)DEC_N << 16); // dim0 hi=0 | tensor_dim1=12288
    g1[3] = (int)((unsigned)KCHUNK << 16); // dim1 hi=0 | tile_dim0=32
    g1[4] = NB_COLS;                    // tile_dim1=512, tile_dim2=0
    g1[5] = HID;                        // tensor_dim0_stride lo = 512
    g1[6] = 0;                          // stride0 hi | tensor_dim1_stride lo
    g1[7] = 0;
    i32x4 z4 = {0, 0, 0, 0};            // groups 2/3 unused (<=2D tile)
    i32x8 z8 = {0, 0, 0, 0, 0, 0, 0, 0};
    __builtin_amdgcn_tensor_load_to_lds(g0, g1, z4, z4, z8, 0);
}

// ------------------------- GEMM2: dec = sigmoid(H @ W2 + b2) ---------------
// Block = 8 waves = one 16-row x 512-col output tile. Per K-step the TDM
// stages the 512x32-half W2t chunk into LDS (double buffered); each wave
// does 4 WMMAs (16x64 slice) with B fragments read from LDS.

__global__ void gemm2(const _Float16* __restrict__ H,   // ROWS x 512
                      const _Float16* __restrict__ W2t, // 12288 x 512 (N-major)
                      const float*    __restrict__ b2,
                      float*          __restrict__ dec) { // ROWS x 12288
    __shared__ __align__(16) _Float16 bufs[2][NB_COLS * KCHUNK]; // 2 x 32KB

    const int tm   = blockIdx.x / (DEC_N / NB_COLS);   // 0..134
    const int nb   = blockIdx.x % (DEC_N / NB_COLS);   // 0..23
    const int lane = threadIdx.x & 31;
    const int wib  = threadIdx.x >> 5;                 // wave in block, 0..7
    const bool w0  = (wib == 0);

    const int row = tm * 16 + (lane & 15);
    const int col = lane & 15;

    // prologue: DMA K-chunk 0 into buffer 0
    if (w0) {
        tdm_load_b_chunk(W2t + (size_t)nb * NB_COLS * HID,
                         (unsigned)(unsigned long long)(&bufs[0][0]));
    }

    v8f acc0 = {}, acc1 = {}, acc2 = {}, acc3 = {};
    for (int kt = 0; kt < HID / KCHUNK; ++kt) {
        const int cur = kt & 1;
        if (w0) __builtin_amdgcn_s_wait_tensorcnt(0);  // buf[cur] landed
        __syncthreads();                                // publish; buf[cur^1] free
        if (w0 && (kt + 1 < HID / KCHUNK)) {            // prefetch next chunk
            tdm_load_b_chunk(W2t + (size_t)nb * NB_COLS * HID + (kt + 1) * KCHUNK,
                             (unsigned)(unsigned long long)(&bufs[cur ^ 1][0]));
        }
        FragT a = load_a_frag(H, row, kt * KCHUNK, HID, lane);
        const _Float16* chunk = &bufs[cur][0];
        FragT b0 = load_b_frag_lds(chunk, wib * 64 + 0  + col, lane);
        FragT b1f = load_b_frag_lds(chunk, wib * 64 + 16 + col, lane);
        FragT b2f = load_b_frag_lds(chunk, wib * 64 + 32 + col, lane);
        FragT b3f = load_b_frag_lds(chunk, wib * 64 + 48 + col, lane);
        acc0 = __builtin_amdgcn_wmma_f32_16x16x32_f16(false, a.h, false, b0.h,
                                                      (short)0, acc0, false, false);
        acc1 = __builtin_amdgcn_wmma_f32_16x16x32_f16(false, a.h, false, b1f.h,
                                                      (short)0, acc1, false, false);
        acc2 = __builtin_amdgcn_wmma_f32_16x16x32_f16(false, a.h, false, b2f.h,
                                                      (short)0, acc2, false, false);
        acc3 = __builtin_amdgcn_wmma_f32_16x16x32_f16(false, a.h, false, b3f.h,
                                                      (short)0, acc3, false, false);
        __syncthreads();                                // done reading buf[cur]
    }

    const int mbase = tm * 16 + ((lane >> 4) << 3);
    v8f accs[4] = {acc0, acc1, acc2, acc3};
#pragma unroll
    for (int j = 0; j < 4; ++j) {
        int n = nb * NB_COLS + wib * 64 + j * 16 + col;
        float bias = b2[n];
#pragma unroll
        for (int r = 0; r < 8; ++r) {
            float t = accs[j][r] + bias;
            dec[(size_t)(mbase + r) * DEC_N + n] = 1.f / (1.f + expf(-t));
        }
    }
}

// ------------------------- fused warp + composite --------------------------

__global__ void composite(const float* __restrict__ dec,     // ROWS x 3 x 64 x 64
                          const float* __restrict__ s_where, // ROWS x 4 (sorted)
                          const int*   __restrict__ s_pres,  // ROWS (sorted)
                          float*       __restrict__ out) {   // B x 3 x 128 x 128
    int pix = blockIdx.x * blockDim.x + threadIdx.x;         // 0 .. B*IMG*IMG-1
    int b = pix >> 14;
    int y = (pix >> 7) & (IMG - 1);
    int x = pix & (IMG - 1);
    float gx = -1.f + (2.f / 127.f) * (float)x;
    float gy = -1.f + (2.f / 127.f) * (float)y;

    float o0 = 0.f, o1 = 0.f, o2 = 0.f;
    bool  f0 = false, f1 = false, f2 = false;

    for (int p = 0; p < N_FEAT; ++p) {
        if (f0 && f1 && f2) break;
        int gi = b * N_FEAT + p;
        if (s_pres[gi] == 0) continue;                       // imgs *= pres
        float cx = s_where[gi * 4 + 0];
        float cy = s_where[gi * 4 + 1];
        float w  = fmaxf(s_where[gi * 4 + 2], 0.01f);
        float h  = fmaxf(s_where[gi * 4 + 3], 0.01f);
        float u  = (gx - (2.f * cx - 1.f)) / w;
        float v  = (gy - (2.f * cy - 1.f)) / h;
        float px = (u + 1.f) * 0.5f * (float)(OBJ - 1);
        float py = (v + 1.f) * 0.5f * (float)(OBJ - 1);
        float x0 = floorf(px), y0 = floorf(py);
        float wx = px - x0,   wy = py - y0;
        int x0i = (int)x0, y0i = (int)y0;

        bool vx0 = (x0i >= 0) && (x0i < OBJ);
        bool vx1 = (x0i + 1 >= 0) && (x0i + 1 < OBJ);
        bool vy0 = (y0i >= 0) && (y0i < OBJ);
        bool vy1 = (y0i + 1 >= 0) && (y0i + 1 < OBJ);
        float w00 = (1.f - wy) * (1.f - wx) * (float)(vy0 && vx0);
        float w01 = (1.f - wy) * wx         * (float)(vy0 && vx1);
        float w10 = wy         * (1.f - wx) * (float)(vy1 && vx0);
        float w11 = wy         * wx         * (float)(vy1 && vx1);
        if (w00 == 0.f && w01 == 0.f && w10 == 0.f && w11 == 0.f) continue;

        int xc0 = min(max(x0i, 0), OBJ - 1);
        int xc1 = min(max(x0i + 1, 0), OBJ - 1);
        int yc0 = min(max(y0i, 0), OBJ - 1);
        int yc1 = min(max(y0i + 1, 0), OBJ - 1);
        const float* img = dec + (size_t)gi * DEC_N;
#pragma unroll
        for (int c = 0; c < 3; ++c) {
            const float* ch = img + c * (OBJ * OBJ);
            float val = ch[yc0 * OBJ + xc0] * w00 + ch[yc0 * OBJ + xc1] * w01 +
                        ch[yc1 * OBJ + xc0] * w10 + ch[yc1 * OBJ + xc1] * w11;
            if (val != 0.f) {
                if (c == 0) { if (!f0) { o0 = val; f0 = true; } }
                if (c == 1) { if (!f1) { o1 = val; f1 = true; } }
                if (c == 2) { if (!f2) { o2 = val; f2 = true; } }
            }
        }
    }
    out[(((size_t)b * 3 + 0) * IMG + y) * IMG + x] = o0;
    out[(((size_t)b * 3 + 1) * IMG + y) * IMG + x] = o1;
    out[(((size_t)b * 3 + 2) * IMG + y) * IMG + x] = o2;
}

// ------------------------- launch ------------------------------------------

static inline size_t align256(size_t x) { return (x + 255) & ~(size_t)255; }

extern "C" void kernel_launch(void* const* d_in, const int* in_sizes, int n_in,
                              void* d_out, int out_size, void* d_ws, size_t ws_size,
                              hipStream_t stream) {
    const float* z_what  = (const float*)d_in[0];  // B x 135 x 64
    const float* z_where = (const float*)d_in[1];  // B x 540 x 4
    const float* z_depth = (const float*)d_in[2];  // B x 135 x 1
    const float* W1      = (const float*)d_in[3];  // 64 x 512
    const float* b1      = (const float*)d_in[4];  // 512
    const float* W2      = (const float*)d_in[5];  // 512 x 12288
    const float* b2      = (const float*)d_in[6];  // 12288
    const int*   z_pres  = (const int*)d_in[7];    // B x 540 x 1
    float* out = (float*)d_out;

    // workspace carve-up
    char* p = (char*)d_ws;
    _Float16* sWhat = (_Float16*)p; p += align256((size_t)ROWS * ZWHAT * 2);
    _Float16* H     = (_Float16*)p; p += align256((size_t)ROWS * HID * 2);
    _Float16* W1t   = (_Float16*)p; p += align256((size_t)HID * ZWHAT * 2);
    _Float16* W2t   = (_Float16*)p; p += align256((size_t)DEC_N * HID * 2);
    float*    sWhere= (float*)p;    p += align256((size_t)ROWS * 4 * 4);
    int*      sPres = (int*)p;      p += align256((size_t)ROWS * 4);
    float*    dec   = (float*)p;    p += align256((size_t)ROWS * DEC_N * 4);
    (void)ws_size; (void)n_in; (void)in_sizes; (void)out_size;

    // K0: weight conversion + transpose to N-major f16
    convW1t<<<(ZWHAT * HID + 255) / 256, 256, 0, stream>>>(W1, W1t);
    convW2t<<<((size_t)HID * DEC_N + 255) / 256, 256, 0, stream>>>(W2, W2t);

    // K1: per-batch stable depth sort + gather
    sort_gather<<<BB, 256, 0, stream>>>(z_what, z_where, z_depth, z_pres,
                                        sWhat, sWhere, sPres);

    // K2: GEMM1 (135x32 tiles, 1 wave each, 8 waves/block)
    gemm1<<<(ROWS / 16) * (HID / 16) / 8, 256, 0, stream>>>(sWhat, W1t, b1, H);

    // K3: GEMM2 (135 x 24 blocks of 16x512, TDM-staged B, 8 waves/block)
    gemm2<<<(ROWS / 16) * (DEC_N / NB_COLS), 256, 0, stream>>>(H, W2t, b2, dec);

    // K4: fused STN warp + z-composite
    composite<<<(BB * IMG * IMG) / 256, 256, 0, stream>>>(dec, sWhere, sPres, out);
}
